// FQ_LISCS_Encoder_52158082843110
// MI455X (gfx1250) — compile-verified, matrix operand order checked by reading, not run on hardware
//
#include <hip/hip_runtime.h>
#include <hip/hip_bf16.h>

typedef __attribute__((ext_vector_type(16))) _Float16 v16h;
typedef __attribute__((ext_vector_type(8)))  float    v8f;
typedef _Float16 half_t;

// K index map for 16-bit WMMA A/B fragments (16x32 / 32x16), per ISA 7.12.2:
//   v<4 : K = 2v + p + 8*hi ;  v>=4 : K = 16 + 2(v-4) + p + 8*hi
__device__ __forceinline__ int kpair(int v, int hi) {
  return (v < 4 ? v * 2 : 16 + (v - 4) * 2) + hi * 8;
}

union h4pack { half_t h[4]; uint2 u; };

#define GLDK 36  // LDS stride in halves (72B): keeps 8B alignment, staggers banks

// ---------------------------------------------------------------------------
// GEMM: C[M,N] = A[M,K] * W[N,K]^T (+ bias[N]); fp32 in/out, f16 WMMA.
// Block = 256 threads (8 waves), tile 128(M) x 64(N).
// Wave tile 32x32 = 4 WMMA per K-step (2 A frags x 2 B frags).
// FAST=true: M%128==0, N%64==0, K%32==0 -> no guards, no branches in K loop.
// ---------------------------------------------------------------------------
template <bool FAST>
__global__ __launch_bounds__(256)
void gemm_wmma(const float* __restrict__ A, const float* __restrict__ W,
               const float* __restrict__ bias, float* __restrict__ C,
               int M, int N, int K)
{
  __shared__ half_t sA[128 * GLDK];
  __shared__ half_t sB[64 * GLDK];

  const int bm = blockIdx.x * 128;
  const int bn = blockIdx.y * 64;
  const int tid = threadIdx.x;
  const int wave = tid >> 5, lane = tid & 31;
  const int wm = wave >> 1;          // 0..3 : 32-row stripe
  const int wn = wave & 1;           // 0..1 : 32-col half
  const int m16 = lane & 15, hi = lane >> 4;

  v8f acc00 = {}, acc01 = {}, acc10 = {}, acc11 = {};

  const int r0 = tid >> 3;         // 0..31
  const int c0 = (tid & 7) * 4;    // 0,4,...,28

  for (int k0 = 0; k0 < K; k0 += 32) {
    // ---- stage A tile (128x32) and W tile (64x32) as f16 ----
    if (FAST) {
#pragma unroll
      for (int ch = 0; ch < 4; ++ch) {
        const int r = r0 + 32 * ch;
        const float4 v = *(const float4*)&A[(size_t)(bm + r) * K + k0 + c0];
        h4pack p;
        p.h[0] = (half_t)v.x; p.h[1] = (half_t)v.y;
        p.h[2] = (half_t)v.z; p.h[3] = (half_t)v.w;
        *(uint2*)&sA[r * GLDK + c0] = p.u;
      }
#pragma unroll
      for (int ch = 0; ch < 2; ++ch) {
        const int r = r0 + 32 * ch;
        const float4 v = *(const float4*)&W[(size_t)(bn + r) * K + k0 + c0];
        h4pack p;
        p.h[0] = (half_t)v.x; p.h[1] = (half_t)v.y;
        p.h[2] = (half_t)v.z; p.h[3] = (half_t)v.w;
        *(uint2*)&sB[r * GLDK + c0] = p.u;
      }
    } else {
      for (int i = tid; i < 128 * 32; i += 256) {
        int r = i >> 5, c = i & 31;
        int gm = bm + r, gk = k0 + c;
        float v = (gm < M && gk < K) ? A[(size_t)gm * K + gk] : 0.0f;
        sA[r * GLDK + c] = (half_t)v;
      }
      for (int i = tid; i < 64 * 32; i += 256) {
        int r = i >> 5, c = i & 31;
        int gn = bn + r, gk = k0 + c;
        float v = (gn < N && gk < K) ? W[(size_t)gn * K + gk] : 0.0f;
        sB[r * GLDK + c] = (half_t)v;
      }
    }
    __syncthreads();

    if (FAST && k0 + 32 < K) {  // speculative prefetch of next A stripe
      __builtin_prefetch(&A[(size_t)(bm + (tid & 127)) * K + k0 + 32], 0, 0);
    }

    union { v16h v; unsigned u[8]; } af0, af1, bf0, bf1;
    const int ar0 = (wm * 32 + m16) * GLDK;
    const int ar1 = (wm * 32 + 16 + m16) * GLDK;
    const int br0 = (wn * 32 + m16) * GLDK;
    const int br1 = (wn * 32 + 16 + m16) * GLDK;
#pragma unroll
    for (int v = 0; v < 8; ++v) {
      const int kk = kpair(v, hi);
      af0.u[v] = *(const unsigned*)&sA[ar0 + kk];
      af1.u[v] = *(const unsigned*)&sA[ar1 + kk];
      bf0.u[v] = *(const unsigned*)&sB[br0 + kk];
      bf1.u[v] = *(const unsigned*)&sB[br1 + kk];
    }
    acc00 = __builtin_amdgcn_wmma_f32_16x16x32_f16(false, af0.v, false, bf0.v,
                                                   (short)0, acc00, false, false);
    acc01 = __builtin_amdgcn_wmma_f32_16x16x32_f16(false, af0.v, false, bf1.v,
                                                   (short)0, acc01, false, false);
    acc10 = __builtin_amdgcn_wmma_f32_16x16x32_f16(false, af1.v, false, bf0.v,
                                                   (short)0, acc10, false, false);
    acc11 = __builtin_amdgcn_wmma_f32_16x16x32_f16(false, af1.v, false, bf1.v,
                                                   (short)0, acc11, false, false);
    __syncthreads();
  }

  // D layout: VGPR r holds (M = r + 8*hi, N = lane&15)
  const int gn0 = bn + wn * 32 + m16;
  const int gn1 = gn0 + 16;
  if (FAST) {
    const float b0 = bias ? bias[gn0] : 0.0f;
    const float b1 = bias ? bias[gn1] : 0.0f;
#pragma unroll
    for (int r = 0; r < 8; ++r) {
      const int gma = bm + wm * 32 + r + 8 * hi;
      const int gmb = gma + 16;
      C[(size_t)gma * N + gn0] = acc00[r] + b0;
      C[(size_t)gma * N + gn1] = acc01[r] + b1;
      C[(size_t)gmb * N + gn0] = acc10[r] + b0;
      C[(size_t)gmb * N + gn1] = acc11[r] + b1;
    }
  } else {
    const float b0 = (bias && gn0 < N) ? bias[gn0] : 0.0f;
    const float b1 = (bias && gn1 < N) ? bias[gn1] : 0.0f;
#pragma unroll
    for (int r = 0; r < 8; ++r) {
      const int gma = bm + wm * 32 + r + 8 * hi;
      const int gmb = gma + 16;
      if (gma < M) {
        if (gn0 < N) C[(size_t)gma * N + gn0] = acc00[r] + b0;
        if (gn1 < N) C[(size_t)gma * N + gn1] = acc01[r] + b1;
      }
      if (gmb < M) {
        if (gn0 < N) C[(size_t)gmb * N + gn0] = acc10[r] + b0;
        if (gn1 < N) C[(size_t)gmb * N + gn1] = acc11[r] + b1;
      }
    }
  }
}

// ---------------------------------------------------------------------------
// Fused window attention: one block per (batch, window, head). N=64, d=32.
// Shift-roll folded into gather/scatter; shift mask computed inline.
// ---------------------------------------------------------------------------
__global__ __launch_bounds__(256)
void attn_wmma(const float* __restrict__ qkv,  // [B, L, 3, heads, 32]
               const float* __restrict__ rpb,  // [225, heads]
               float* __restrict__ out,        // [B, L, heads*32]
               int Himg, int Wimg, int heads, int shift)
{
  const int C = heads * 32;
  const int L = Himg * Wimg;
  const int nWx = Wimg >> 3, nWy = Himg >> 3;
  int id = blockIdx.x;
  const int head = id % heads; id /= heads;
  const int wx = id % nWx;     id /= nWx;
  const int wy = id % nWy;     id /= nWy;
  const int b = id;

  __shared__ half_t sQ[64 * 36], sK[64 * 36], sV[64 * 36];
  __shared__ float  sS[64 * 68];
  __shared__ half_t sP[64 * 72];
  __shared__ float  rbuf[64];

  const int tid = threadIdx.x;
  const float qscale = 0.17677669529663687f;  // 32^-0.5

  // vectorized QKV gather: 64 rows x 8 float4s per matrix
  for (int idx = tid; idx < 64 * 8; idx += 256) {
    const int n = idx >> 3, d0 = (idx & 7) * 4;
    const int i = n >> 3, j = n & 7;
    const int y = (wy * 8 + i + shift) % Himg;
    const int x = (wx * 8 + j + shift) % Wimg;
    const size_t base = ((size_t)(b * L + y * Wimg + x) * 3) * C + head * 32 + d0;
    const float4 q = *(const float4*)&qkv[base];
    const float4 k = *(const float4*)&qkv[base + C];
    const float4 v = *(const float4*)&qkv[base + 2 * C];
    h4pack pq, pk, pv;
    pq.h[0] = (half_t)(q.x * qscale); pq.h[1] = (half_t)(q.y * qscale);
    pq.h[2] = (half_t)(q.z * qscale); pq.h[3] = (half_t)(q.w * qscale);
    pk.h[0] = (half_t)k.x; pk.h[1] = (half_t)k.y;
    pk.h[2] = (half_t)k.z; pk.h[3] = (half_t)k.w;
    pv.h[0] = (half_t)v.x; pv.h[1] = (half_t)v.y;
    pv.h[2] = (half_t)v.z; pv.h[3] = (half_t)v.w;
    *(uint2*)&sQ[n * 36 + d0] = pq.u;
    *(uint2*)&sK[n * 36 + d0] = pk.u;
    *(uint2*)&sV[n * 36 + d0] = pv.u;
  }
  __syncthreads();

  const int wave = tid >> 5, lane = tid & 31;
  const int m16 = lane & 15, hi = lane >> 4;

  // ---- S = Q K^T : 16 tiles (4x4), 2 per wave, single K-step of 32 ----
  {
    const int tm = wave >> 1;
    union { v16h v; unsigned u[8]; } af;
#pragma unroll
    for (int v = 0; v < 8; ++v)
      af.u[v] = *(const unsigned*)&sQ[(tm * 16 + m16) * 36 + kpair(v, hi)];
#pragma unroll
    for (int t = 0; t < 2; ++t) {
      const int tn = (wave & 1) * 2 + t;
      union { v16h v; unsigned u[8]; } bf;
#pragma unroll
      for (int v = 0; v < 8; ++v)
        bf.u[v] = *(const unsigned*)&sK[(tn * 16 + m16) * 36 + kpair(v, hi)];
      v8f acc = {};
      acc = __builtin_amdgcn_wmma_f32_16x16x32_f16(false, af.v, false, bf.v,
                                                   (short)0, acc, false, false);
#pragma unroll
      for (int r = 0; r < 8; ++r)
        sS[(tm * 16 + r + 8 * hi) * 68 + tn * 16 + m16] = acc[r];
    }
  }
  __syncthreads();

  // ---- softmax with relative position bias + inline shift mask ----
  if (tid < 64) {
    const int i = tid >> 3, j = tid & 7;
    int cnt_r = 0;
    if (shift > 0) {
      int ys = wy * 8 + i, xs = wx * 8 + j;
      int ry = ys < Himg - 8 ? 0 : (ys < Himg - 4 ? 1 : 2);
      int rx = xs < Wimg - 8 ? 0 : (xs < Wimg - 4 ? 1 : 2);
      cnt_r = ry * 3 + rx;
    }
    float mx = -1e30f;
    for (int m = 0; m < 64; ++m) {
      int im = m >> 3, jm = m & 7;
      float v = sS[tid * 68 + m];
      v += rpb[((i - im + 7) * 15 + (j - jm + 7)) * heads + head];
      if (shift > 0) {
        int ysm = wy * 8 + im, xsm = wx * 8 + jm;
        int rym = ysm < Himg - 8 ? 0 : (ysm < Himg - 4 ? 1 : 2);
        int rxm = xsm < Wimg - 8 ? 0 : (xsm < Wimg - 4 ? 1 : 2);
        if (rym * 3 + rxm != cnt_r) v -= 100.0f;
      }
      sS[tid * 68 + m] = v;
      mx = fmaxf(mx, v);
    }
    float sum = 0.0f;
    for (int m = 0; m < 64; ++m) {
      float e = __expf(sS[tid * 68 + m] - mx);
      sS[tid * 68 + m] = e; sum += e;
    }
    float inv = 1.0f / sum, rmax = 0.0f;
    for (int m = 0; m < 64; ++m) {
      float p = sS[tid * 68 + m] * inv;
      sS[tid * 68 + m] = p; rmax = fmaxf(rmax, p);
    }
    rbuf[tid] = rmax;
  }
  __syncthreads();
  if (tid == 0) {  // per-tile scale (local approximation of per-tensor fq)
    float m = 0.0f;
    for (int r = 0; r < 64; ++r) m = fmaxf(m, rbuf[r]);
    rbuf[0] = fmaxf(m, 1e-5f) / 127.0f;
  }
  __syncthreads();
  const float ps = rbuf[0];
  const float psi = 1.0f / ps;
  for (int idx = tid; idx < 64 * 64; idx += 256) {
    int r = idx >> 6, c2 = idx & 63;
    float q = rintf(sS[r * 68 + c2] * psi);
    q = fminf(fmaxf(q, -127.0f), 127.0f);
    sP[r * 72 + c2] = (half_t)(q * ps);
  }
  __syncthreads();

  // ---- O = P * V : 8 tiles (4x2), 1 per wave, two K-steps of 32 ----
  {
    const int tm = wave >> 1, tn = wave & 1;
    v8f acc = {};
#pragma unroll
    for (int ks = 0; ks < 2; ++ks) {
      union { v16h v; unsigned u[8]; } af;
#pragma unroll
      for (int v = 0; v < 8; ++v)
        af.u[v] = *(const unsigned*)&sP[(tm * 16 + m16) * 72 + ks * 32 + kpair(v, hi)];
      union { v16h v; half_t h[16]; } bf;
#pragma unroll
      for (int e = 0; e < 16; ++e) {
        int v = e >> 1;
        int kk = ks * 32 + kpair(v, hi) + (e & 1);
        bf.h[e] = sV[kk * 36 + tn * 16 + m16];
      }
      acc = __builtin_amdgcn_wmma_f32_16x16x32_f16(false, af.v, false, bf.v,
                                                   (short)0, acc, false, false);
    }
#pragma unroll
    for (int r = 0; r < 8; ++r) {
      int n = tm * 16 + r + 8 * hi;
      int i = n >> 3, j = n & 7;
      int y = (wy * 8 + i + shift) % Himg;
      int x = (wx * 8 + j + shift) % Wimg;
      out[(size_t)(b * L + y * Wimg + x) * C + head * 32 + tn * 16 + m16] = acc[r];
    }
  }
}

// ---------------------------------------------------------------------------
// Elementwise / reduction kernels
// ---------------------------------------------------------------------------
__global__ __launch_bounds__(256)
void ln_kernel(const float* __restrict__ x, const float* __restrict__ g,
               const float* __restrict__ bta, float* __restrict__ y, int C)
{
  const size_t row = blockIdx.x;
  const float* xr = x + row * (size_t)C;
  float* yr = y + row * (size_t)C;
  float s = 0.0f, s2 = 0.0f;
  for (int c = threadIdx.x; c < C; c += blockDim.x) {
    float v = xr[c]; s += v; s2 += v * v;
  }
#pragma unroll
  for (int o = 16; o > 0; o >>= 1) { s += __shfl_xor(s, o, 32); s2 += __shfl_xor(s2, o, 32); }
  __shared__ float as[8], as2[8];
  const int w = threadIdx.x >> 5, l = threadIdx.x & 31;
  if (l == 0) { as[w] = s; as2[w] = s2; }
  __syncthreads();
  if (w == 0) {
    s  = (l < 8) ? as[l]  : 0.0f;
    s2 = (l < 8) ? as2[l] : 0.0f;
#pragma unroll
    for (int o = 4; o > 0; o >>= 1) { s += __shfl_xor(s, o, 32); s2 += __shfl_xor(s2, o, 32); }
    if (l == 0) { as[0] = s; as2[0] = s2; }
  }
  __syncthreads();
  const float mean = as[0] / C;
  const float var  = as2[0] / C - mean * mean;
  const float rstd = rsqrtf(var + 1e-5f);
  for (int c = threadIdx.x; c < C; c += blockDim.x)
    yr[c] = (xr[c] - mean) * rstd * g[c] + bta[c];
}

__global__ void absmax_kernel(const float* __restrict__ x, size_t n, unsigned* out) {
  float m = 0.0f;
  for (size_t i = blockIdx.x * (size_t)blockDim.x + threadIdx.x; i < n;
       i += (size_t)gridDim.x * blockDim.x)
    m = fmaxf(m, fabsf(x[i]));
#pragma unroll
  for (int o = 16; o > 0; o >>= 1) m = fmaxf(m, __shfl_xor(m, o, 32));
  if ((threadIdx.x & 31) == 0) atomicMax(out, __float_as_uint(m));
}

__global__ void fq_apply_kernel(float* x, const unsigned* amax, size_t n, float lev) {
  const float a = __uint_as_float(*amax);
  const float s = fmaxf(a, 1e-5f) / lev;
  const float inv = 1.0f / s;
  for (size_t i = blockIdx.x * (size_t)blockDim.x + threadIdx.x; i < n;
       i += (size_t)gridDim.x * blockDim.x) {
    float q = rintf(x[i] * inv);
    q = fminf(fmaxf(q, -lev), lev);
    x[i] = q * s;
  }
}

__global__ void add_kernel(const float* a, const float* b, float* o, size_t n) {
  for (size_t i = blockIdx.x * (size_t)blockDim.x + threadIdx.x; i < n;
       i += (size_t)gridDim.x * blockDim.x) o[i] = a[i] + b[i];
}
__global__ void mul_kernel(const float* a, const float* b, float* o, size_t n) {
  for (size_t i = blockIdx.x * (size_t)blockDim.x + threadIdx.x; i < n;
       i += (size_t)gridDim.x * blockDim.x) o[i] = a[i] * b[i];
}
__global__ void gelu_kernel(float* x, size_t n) {
  for (size_t i = blockIdx.x * (size_t)blockDim.x + threadIdx.x; i < n;
       i += (size_t)gridDim.x * blockDim.x) {
    float v = x[i];
    x[i] = 0.5f * v * (1.0f + tanhf(0.7978845608f * (v + 0.044715f * v * v * v)));
  }
}
__global__ void relu_kernel(float* x, size_t n) {
  for (size_t i = blockIdx.x * (size_t)blockDim.x + threadIdx.x; i < n;
       i += (size_t)gridDim.x * blockDim.x) x[i] = fmaxf(x[i], 0.0f);
}
__global__ void sigmoid_kernel(float* x, size_t n) {
  for (size_t i = blockIdx.x * (size_t)blockDim.x + threadIdx.x; i < n;
       i += (size_t)gridDim.x * blockDim.x) x[i] = 1.0f / (1.0f + __expf(-x[i]));
}
__global__ void mul_bcast_kernel(const float* t, const float* bm, float* o,
                                 int rowsPerB, int C, size_t n) {
  for (size_t i = blockIdx.x * (size_t)blockDim.x + threadIdx.x; i < n;
       i += (size_t)gridDim.x * blockDim.x) {
    int c = (int)(i % C);
    int b = (int)(i / ((size_t)rowsPerB * C));
    o[i] = t[i] * bm[b * C + c];
  }
}

// patch-merge gather: out[b, y2*W2+x2, g*C+c]
__global__ void pm_gather_kernel(const float* __restrict__ h, float* __restrict__ o,
                                 int H2, int W2, int Cin, size_t n) {
  for (size_t i = blockIdx.x * (size_t)blockDim.x + threadIdx.x; i < n;
       i += (size_t)gridDim.x * blockDim.x) {
    size_t rem = i;
    int c = (int)(rem % Cin); rem /= Cin;
    int g = (int)(rem % 4);   rem /= 4;
    int x2 = (int)(rem % W2); rem /= W2;
    int y2 = (int)(rem % H2); rem /= H2;
    int b = (int)rem;
    int y = 2 * y2 + ((g == 1) || (g == 3));
    int x = 2 * x2 + (g >= 2);
    o[i] = h[(((size_t)b * (2 * H2) + y) * (2 * W2) + x) * Cin + c];
  }
}

// patch embed: 3->128 conv 2x2 stride 2; out [B, 128*128, 128]
__global__ __launch_bounds__(128)
void patch_embed_kernel(const float* __restrict__ x, const float* __restrict__ w,
                        const float* __restrict__ bias, float* __restrict__ o) {
  const int oc = threadIdx.x;
  int p = blockIdx.x;
  const int ox = p & 127; p >>= 7;
  const int oy = p & 127; p >>= 7;
  const int b = p;
  float acc = bias[oc];
#pragma unroll
  for (int ci = 0; ci < 3; ++ci)
#pragma unroll
    for (int ky = 0; ky < 2; ++ky)
#pragma unroll
      for (int kx = 0; kx < 2; ++kx)
        acc += x[(((size_t)b * 3 + ci) * 256 + 2 * oy + ky) * 256 + 2 * ox + kx] *
               w[((oc * 3 + ci) * 2 + ky) * 2 + kx];
  o[((size_t)b * 16384 + oy * 128 + ox) * 128 + oc] = acc;
}

// ---------------------------------------------------------------------------
// Host orchestration
// ---------------------------------------------------------------------------
extern "C" void kernel_launch(void* const* d_in, const int* in_sizes, int n_in,
                              void* d_out, int out_size, void* d_ws, size_t ws_size,
                              hipStream_t stream)
{
  (void)in_sizes; (void)n_in; (void)out_size; (void)ws_size;
  static const int EMBEDC[4]  = {128, 192, 256, 320};
  static const int DEPTHSC[4] = {2, 2, 6, 2};
  static const int HEADSC[4]  = {4, 6, 8, 10};
  static const int layer_base[4] = {42, 68, 97, 178};  // leaf offsets into params
  const int PE = 207, SM = 211;

  const float* xin = (const float*)d_in[0];
  const float* snr = (const float*)d_in[1];
  auto P = [&](int i) { return (const float*)d_in[2 + i]; };

  // workspace carve (floats)
  float* W = (float*)d_ws;
  size_t off = 0;
  auto carve = [&](size_t nfl) { float* p = W + off; off += nfl; return p; };
  float* bufA = carve((size_t)2 * 16384 * 128);
  float* bufB = carve((size_t)2 * 16384 * 128);
  float* bufC = carve((size_t)2 * 16384 * 512);
  float* bufD = carve((size_t)2 * 16384 * 128);
  float* bufE = carve((size_t)2 * 16384 * 128);
  float* bufT  = carve((size_t)512 * 480);
  float* bufT2 = carve((size_t)512 * 480);
  float* bufM1 = carve(2 * 480);
  float* bufM2 = carve(2 * 480);
  unsigned* amax = (unsigned*)carve(16);

  auto ewg = [&](size_t n) {
    size_t g = (n + 255) / 256; if (g > 65535) g = 65535; return dim3((unsigned)g);
  };
  auto fq = [&](float* buf, size_t n, int bits) {
    hipMemsetAsync(amax, 0, 4, stream);
    absmax_kernel<<<dim3(1024), dim3(256), 0, stream>>>(buf, n, amax);
    float lev = (float)((1 << (bits - 1)) - 1);
    fq_apply_kernel<<<ewg(n), dim3(256), 0, stream>>>(buf, amax, n, lev);
  };
  auto gemm = [&](const float* A, const float* Wt, const float* bias, float* Cp,
                  int M, int N, int K) {
    dim3 g((M + 127) / 128, (N + 63) / 64);
    if ((M % 128 == 0) && (N % 64 == 0) && (K % 32 == 0))
      gemm_wmma<true><<<g, 256, 0, stream>>>(A, Wt, bias, Cp, M, N, K);
    else
      gemm_wmma<false><<<g, 256, 0, stream>>>(A, Wt, bias, Cp, M, N, K);
  };
  auto ln = [&](const float* xi, const float* g, const float* b, float* y,
                int rows, int C) {
    ln_kernel<<<dim3(rows), dim3(256), 0, stream>>>(xi, g, b, y, C);
  };

  // ---- patch embed + LN + fq8 ----
  patch_embed_kernel<<<dim3(2 * 128 * 128), dim3(128), 0, stream>>>(
      xin, P(PE + 3), P(PE + 0), bufA);
  ln(bufA, P(PE + 2), P(PE + 1), bufA, 2 * 16384, 128);
  fq(bufA, (size_t)2 * 16384 * 128, 8);

  // ---- stages ----
  int res = 128;
  for (int l = 0; l < 4; ++l) {
    const int dim = EMBEDC[l], heads = HEADSC[l];
    if (l > 0) {
      const int din = EMBEDC[l - 1];
      const int H2 = res / 2;
      const int mg = layer_base[l] + DEPTHSC[l] * 13;
      const size_t ng = (size_t)2 * H2 * H2 * 4 * din;
      pm_gather_kernel<<<ewg(ng), dim3(256), 0, stream>>>(bufA, bufB, H2, H2, din, ng);
      ln(bufB, P(mg + 1), P(mg + 0), bufB, 2 * H2 * H2, 4 * din);
      gemm(bufB, P(mg + 2), nullptr, bufA, 2 * H2 * H2, dim, 4 * din);
      fq(bufA, (size_t)2 * H2 * H2 * dim, 8);
      res = H2;
    }
    const int L = res * res;
    const int rows = 2 * L;
    const size_t nD = (size_t)rows * dim;
    for (int bi = 0; bi < DEPTHSC[l]; ++bi) {
      const int shift = (bi & 1) ? 4 : 0;
      const int bb = layer_base[l] + bi * 13;
      // leaves: 0 fc1_b 1 fc1_w 2 fc2_b 3 fc2_w 4 n1_b 5 n1_g 6 n2_b 7 n2_g
      //         8 proj_b 9 proj_w 10 qkv_b 11 qkv_w 12 rpb
      ln(bufA, P(bb + 5), P(bb + 4), bufB, rows, dim);
      fq(bufB, nD, 8);
      gemm(bufB, P(bb + 11), P(bb + 10), bufC, rows, 3 * dim, dim);
      fq(bufC, nD * 3, 8);
      attn_wmma<<<dim3(2 * (res / 8) * (res / 8) * heads), dim3(256), 0, stream>>>(
          bufC, P(bb + 12), bufD, res, res, heads, shift);
      gemm(bufD, P(bb + 9), P(bb + 8), bufB, rows, dim, dim);
      fq(bufB, nD, 8);
      add_kernel<<<ewg(nD), dim3(256), 0, stream>>>(bufA, bufB, bufE, nD);
      fq(bufE, nD, 16);
      ln(bufE, P(bb + 7), P(bb + 6), bufB, rows, dim);
      fq(bufB, nD, 8);
      gemm(bufB, P(bb + 1), P(bb + 0), bufC, rows, 4 * dim, dim);
      gelu_kernel<<<ewg(nD * 4), dim3(256), 0, stream>>>(bufC, nD * 4);
      fq(bufC, nD * 4, 8);
      gemm(bufC, P(bb + 3), P(bb + 2), bufD, rows, dim, 4 * dim);
      add_kernel<<<ewg(nD), dim3(256), 0, stream>>>(bufE, bufD, bufA, nD);
      fq(bufA, nD, 16);
    }
  }

  // ---- SNR-adaptive modulation (7 sm layers + modulators) ----
  const size_t nT = (size_t)512 * 480;
  for (int i = 0; i < 7; ++i) {
    const float* inp = (i == 0) ? bufA : bufT2;
    const int Kin = (i == 0) ? 320 : 480;
    gemm(inp, P(SM + 2 * i + 1), P(SM + 2 * i), bufT, 512, 480, Kin);
    fq(bufT, nT, 8);
    const int mb = i * 6;  // b1 b2 b3 w1 w2 w3
    gemm(snr, P(mb + 3), P(mb + 0), bufM1, 2, 480, 1);
    relu_kernel<<<ewg(960), dim3(256), 0, stream>>>(bufM1, 960);
    fq(bufM1, 960, 8);
    gemm(bufM1, P(mb + 4), P(mb + 1), bufM2, 2, 480, 480);
    relu_kernel<<<ewg(960), dim3(256), 0, stream>>>(bufM2, 960);
    fq(bufM2, 960, 8);
    gemm(bufM2, P(mb + 5), P(mb + 2), bufM1, 2, 480, 480);
    fq(bufM1, 960, 8);
    sigmoid_kernel<<<ewg(960), dim3(256), 0, stream>>>(bufM1, 960);
    fq(bufM1, 960, 8);
    mul_bcast_kernel<<<ewg(nT), dim3(256), 0, stream>>>(bufT, bufM1, bufT2, 256, 480, nT);
  }
  const size_t nO = (size_t)512 * 320;
  gemm(bufT2, P(SM + 15), P(SM + 14), bufD, 512, 320, 480);
  fq(bufD, nO, 8);
  sigmoid_kernel<<<ewg(nO), dim3(256), 0, stream>>>(bufD, nO);
  mul_kernel<<<ewg(nO), dim3(256), 0, stream>>>(bufA, bufD, (float*)d_out, nO);
}
